// CausalAttention_5188320493941
// MI455X (gfx1250) — matrix operand order repaired and checked by reading, not compiled, not run on hardware
//
#include <hip/hip_runtime.h>
#include <hip/hip_bf16.h>
#include <math.h>

#define BATCH  2
#define NSEQ   1536
#define DMODEL 1024
#define HEADS  16
#define DHEAD  64
#define MROWS  (BATCH * NSEQ)          // 3072
#define SCALE  0.125f                  // DH^-0.5

typedef __attribute__((ext_vector_type(16))) __bf16 v16bf;
typedef __attribute__((ext_vector_type(8)))  __bf16 v8bf;
typedef __attribute__((ext_vector_type(8)))  float  v8f;

static __device__ __forceinline__ v8f wmma_bf16(v16bf a, v16bf b, v8f c) {
  return __builtin_amdgcn_wmma_f32_16x16x32_bf16(false, a, false, b, (short)0, c, false, false);
}

// A fragment (16x32 bf16): lane = M (mod 16), kgroup = lane>>4.
// p must point at A[m][k0 + kgroup*16]; 16 contiguous K elements.
static __device__ __forceinline__ v16bf load_a_frag(const __bf16* __restrict__ p) {
  v8bf lo = *(const v8bf*)p;
  v8bf hi = *(const v8bf*)(p + 8);
  v16bf r;
#pragma unroll
  for (int i = 0; i < 8; ++i) { r[i] = lo[i]; r[8 + i] = hi[i]; }
  return r;
}

// ---------------------------------------------------------------------------
// Elementwise converts
// ---------------------------------------------------------------------------
__global__ void cvt_f32_to_bf16(const float* __restrict__ in, __bf16* __restrict__ out, size_t n) {
  size_t i = (size_t)blockIdx.x * blockDim.x + threadIdx.x;
  if (i < n) out[i] = (__bf16)in[i];
}

// WT[n][k] = (bf16) W[k][n]  (coalesced writes)
__global__ void transpose_w_bf16(const float* __restrict__ W, __bf16* __restrict__ WT) {
  size_t o = (size_t)blockIdx.x * blockDim.x + threadIdx.x;   // over D*D
  int n = (int)(o / DMODEL);
  int k = (int)(o % DMODEL);
  WT[o] = (__bf16)W[(size_t)k * DMODEL + n];
}

__global__ void zero_f32(float* __restrict__ p, int n) {
  int i = blockIdx.x * blockDim.x + threadIdx.x;
  if (i < n) p[i] = 0.0f;
}

__global__ void gate_finalize(const float* __restrict__ gacc, const float* __restrict__ bg2,
                              float* __restrict__ gate, int n) {
  int i = blockIdx.x * blockDim.x + threadIdx.x;
  if (i < n) gate[i] = 1.0f / (1.0f + expf(-(gacc[i] + bg2[0])));
}

// ---------------------------------------------------------------------------
// bf16 WMMA GEMM: C[M][Nc] = A[M][K] * BT[Nc][K]^T + bias
// Wave tile 32x32 (2x2 accumulators, 4 WMMAs / K-step, 2x operand reuse).
// Block = 8 waves (2 row x 4 col) covering 64(M) x 128(N).
// Epilogues:
//   0: store bf16 row-major (Q/K projections)
//   1: store bf16 transposed into Vt[b][h][dh][token]
//   2: GELU -> dot with wg2 column weights -> atomic into gacc[m]
//   3: store fp32 row-major (final output projection)
// ---------------------------------------------------------------------------
template <int EP>
__global__ __launch_bounds__(256) void gemm_bf16(
    const __bf16* __restrict__ A, const __bf16* __restrict__ BT,
    const float* __restrict__ bias, void* __restrict__ outp,
    const float* __restrict__ wg2, float* __restrict__ gacc, int K) {
  const int lane = threadIdx.x & 31;
  const int wave = threadIdx.x >> 5;
  const int l16  = lane & 15;
  const int half = lane >> 4;
  const int m0 = blockIdx.y * 64  + (wave & 1) * 32;
  const int n0 = blockIdx.x * 128 + (wave >> 1) * 32;

  const __bf16* arow0 = A  + (size_t)(m0 + l16) * K + half * 16;
  const __bf16* arow1 = arow0 + (size_t)16 * K;
  const __bf16* brow0 = BT + (size_t)(n0 + l16) * K + half * 16;
  const __bf16* brow1 = brow0 + (size_t)16 * K;

  v8f acc[2][2] = {{v8f{}, v8f{}}, {v8f{}, v8f{}}};
#pragma unroll 2
  for (int k = 0; k < K; k += 32) {
    // keep the WGP$ primed ~8 K-steps ahead (global_prefetch_b8)
    __builtin_prefetch(arow0 + k + 256, 0, 1);
    __builtin_prefetch(arow1 + k + 256, 0, 1);
    __builtin_prefetch(brow0 + k + 256, 0, 1);
    __builtin_prefetch(brow1 + k + 256, 0, 1);
    v16bf a0 = load_a_frag(arow0 + k);
    v16bf a1 = load_a_frag(arow1 + k);
    v16bf b0 = *(const v16bf*)(brow0 + k);   // contiguous 16 K-elems per lane
    v16bf b1 = *(const v16bf*)(brow1 + k);
    acc[0][0] = wmma_bf16(a0, b0, acc[0][0]);
    acc[0][1] = wmma_bf16(a0, b1, acc[0][1]);
    acc[1][0] = wmma_bf16(a1, b0, acc[1][0]);
    acc[1][1] = wmma_bf16(a1, b1, acc[1][1]);
  }

  if constexpr (EP == 2) {                 // GELU(x)*wg2[n] partial dot per row
#pragma unroll
    for (int i = 0; i < 2; ++i) {
      float part[8];
#pragma unroll
      for (int v = 0; v < 8; ++v) part[v] = 0.0f;
#pragma unroll
      for (int j = 0; j < 2; ++j) {
        int n = n0 + j * 16 + l16;
        float bn = bias[n];
        float w  = wg2[n];
#pragma unroll
        for (int v = 0; v < 8; ++v) {
          float t = acc[i][j][v] + bn;
          float g = 0.5f * t * (1.0f + erff(t * 0.70710678118f));
          part[v] += g * w;
        }
      }
#pragma unroll
      for (int mask = 1; mask < 16; mask <<= 1)
#pragma unroll
        for (int v = 0; v < 8; ++v) part[v] += __shfl_xor(part[v], mask, 32);
      if (l16 == 0) {
#pragma unroll
        for (int v = 0; v < 8; ++v)
          atomicAdd(&gacc[m0 + i * 16 + half * 8 + v], part[v]);
      }
    }
  } else {
#pragma unroll
    for (int i = 0; i < 2; ++i) {
#pragma unroll
      for (int j = 0; j < 2; ++j) {
        const int n = n0 + j * 16 + l16;
        const float bn = bias[n];
        if constexpr (EP == 0) {           // bf16 row-major
          __bf16* out = (__bf16*)outp;
#pragma unroll
          for (int v = 0; v < 8; ++v) {
            int m = m0 + i * 16 + half * 8 + v;
            out[(size_t)m * DMODEL + n] = (__bf16)(acc[i][j][v] + bn);
          }
        } else if constexpr (EP == 1) {    // Vt[(b*H+h)*DH+dh][token]
          __bf16* out = (__bf16*)outp;
          int h = n >> 6, dh = n & 63;
          int mbase = m0 + i * 16 + half * 8;
          int b = mbase / NSEQ, tok0 = mbase % NSEQ;
          v8bf t;
#pragma unroll
          for (int v = 0; v < 8; ++v) t[v] = (__bf16)(acc[i][j][v] + bn);
          *(v8bf*)(out + ((size_t)((b * HEADS + h) * DHEAD + dh)) * NSEQ + tok0) = t;
        } else {                           // fp32 row-major (d_out)
          float* out = (float*)outp;
#pragma unroll
          for (int v = 0; v < 8; ++v) {
            int m = m0 + i * 16 + half * 8 + v;
            out[(size_t)m * DMODEL + n] = acc[i][j][v] + bn;
          }
        }
      }
    }
  }
}

// ---------------------------------------------------------------------------
// Flash-style causal attention + entropy + gate scaling.
// One wave per 16-query tile of one (b, h); 4 waves / block.
// Qb, Kb: bf16 [B*N][D] (head h at cols h*64..); Vt: bf16 [B*H*DH][N].
// ---------------------------------------------------------------------------
__global__ __launch_bounds__(128) void attn_kernel(
    const __bf16* __restrict__ Qb, const __bf16* __restrict__ Kb,
    const __bf16* __restrict__ Vt, const float* __restrict__ gate,
    __bf16* __restrict__ Oc, float* __restrict__ entOut) {
  __shared__ __align__(32) __bf16 Plds[4][16 * 32];

  const int lane = threadIdx.x & 31;
  const int wave = threadIdx.x >> 5;
  const int l16  = lane & 15;
  const int half = lane >> 4;
  const int q0 = (blockIdx.x * 4 + wave) * 16;
  const int b  = blockIdx.y >> 4;
  const int h  = blockIdx.y & 15;

  const __bf16* Qbase = Qb + ((size_t)(b * NSEQ) + q0) * DMODEL + h * DHEAD;
  const __bf16* Kbase = Kb + (size_t)(b * NSEQ) * DMODEL + h * DHEAD;
  const __bf16* Vbase = Vt + (size_t)((b * HEADS + h) * DHEAD) * NSEQ;

  // Q fragments for d = 0..31 and 32..63
  v16bf qa[2];
  {
    const __bf16* ar = Qbase + (size_t)l16 * DMODEL + half * 16;
    qa[0] = load_a_frag(ar);
    qa[1] = load_a_frag(ar + 32);
  }

  float mrow[8], lrow[8], trow[8];
  v8f accO[4] = {v8f{}, v8f{}, v8f{}, v8f{}};
#pragma unroll
  for (int v = 0; v < 8; ++v) { mrow[v] = -1e30f; lrow[v] = 0.0f; trow[v] = 0.0f; }

  const int qlast = q0 + 15;
  for (int k0 = 0; k0 <= qlast; k0 += 32) {
    // scores for two 16-key subtiles (K-dim = DH=64 -> 2 WMMAs each)
    float zt[2][8];
#pragma unroll
    for (int kt = 0; kt < 2; ++kt) {
      const __bf16* br = Kbase + (size_t)(k0 + kt * 16 + l16) * DMODEL + half * 16;
      __builtin_prefetch(br + 32 * DMODEL, 0, 1);     // next 32-key tile of K
      v16bf b0 = *(const v16bf*)br;
      v16bf b1 = *(const v16bf*)(br + 32);
      v8f s = {};
      s = wmma_bf16(qa[0], b0, s);
      s = wmma_bf16(qa[1], b1, s);
      int key = k0 + kt * 16 + l16;
#pragma unroll
      for (int v = 0; v < 8; ++v) {
        int qr = q0 + half * 8 + v;
        zt[kt][v] = s[v] * SCALE + (key > qr ? -1e9f : 0.0f);
      }
    }
    // tile row-max across 32 keys (16-lane reduction per half)
    float tmax[8];
#pragma unroll
    for (int v = 0; v < 8; ++v) tmax[v] = fmaxf(zt[0][v], zt[1][v]);
#pragma unroll
    for (int mask = 1; mask < 16; mask <<= 1)
#pragma unroll
      for (int v = 0; v < 8; ++v) tmax[v] = fmaxf(tmax[v], __shfl_xor(tmax[v], mask, 32));
    // rescale running state
#pragma unroll
    for (int v = 0; v < 8; ++v) {
      float mnew = fmaxf(mrow[v], tmax[v]);
      float f = __expf(mrow[v] - mnew);
      mrow[v] = mnew; lrow[v] *= f; trow[v] *= f;
#pragma unroll
      for (int g = 0; g < 4; ++g) accO[g][v] *= f;
    }
    // probabilities -> LDS (bf16), plus l and <z> partials
    float lp[8], tp[8];
#pragma unroll
    for (int v = 0; v < 8; ++v) { lp[v] = 0.0f; tp[v] = 0.0f; }
#pragma unroll
    for (int kt = 0; kt < 2; ++kt)
#pragma unroll
      for (int v = 0; v < 8; ++v) {
        float p = __expf(zt[kt][v] - mrow[v]);
        lp[v] += p; tp[v] += p * zt[kt][v];
        Plds[wave][(half * 8 + v) * 32 + kt * 16 + l16] = (__bf16)p;
      }
#pragma unroll
    for (int mask = 1; mask < 16; mask <<= 1)
#pragma unroll
      for (int v = 0; v < 8; ++v) {
        lp[v] += __shfl_xor(lp[v], mask, 32);
        tp[v] += __shfl_xor(tp[v], mask, 32);
      }
#pragma unroll
    for (int v = 0; v < 8; ++v) { lrow[v] += lp[v]; trow[v] += tp[v]; }

    // P(16x32) @ V(32x16 per dh group): A from LDS, B from transposed V
    v16bf pa = load_a_frag(&Plds[wave][l16 * 32 + half * 16]);
#pragma unroll
    for (int g = 0; g < 4; ++g) {
      const __bf16* vr = Vbase + (size_t)(g * 16 + l16) * NSEQ + k0 + half * 16;
      __builtin_prefetch(vr + 32, 0, 1);              // next 32-key tile of V^T
      v16bf vb = *(const v16bf*)vr;
      accO[g] = wmma_bf16(pa, vb, accO[g]);
    }
  }

  // finalize: gate/l normalization, entropy = lse - <z>
  float gv[8], ent[8];
#pragma unroll
  for (int v = 0; v < 8; ++v) {
    int qr = q0 + half * 8 + v;
    float inv = 1.0f / lrow[v];
    ent[v] = mrow[v] + logf(lrow[v]) - trow[v] * inv;
    gv[v] = gate[b * NSEQ + qr] * inv;
  }
#pragma unroll
  for (int g = 0; g < 4; ++g) {
    int n = h * DHEAD + g * 16 + l16;
#pragma unroll
    for (int v = 0; v < 8; ++v) {
      int m = b * NSEQ + q0 + half * 8 + v;
      Oc[(size_t)m * DMODEL + n] = (__bf16)(accO[g][v] * gv[v]);
    }
  }
  if (l16 == 0) {
#pragma unroll
    for (int v = 0; v < 8; ++v)
      entOut[(size_t)(b * HEADS + h) * NSEQ + q0 + half * 8 + v] = ent[v];
  }
}

// ---------------------------------------------------------------------------
extern "C" void kernel_launch(void* const* d_in, const int* in_sizes, int n_in,
                              void* d_out, int out_size, void* d_ws, size_t ws_size,
                              hipStream_t stream) {
  (void)in_sizes; (void)n_in; (void)out_size; (void)ws_size;
  const float* x   = (const float*)d_in[0];
  // d_in[1] = attn_bias: exact causal tril mask, applied analytically in-kernel
  const float* Wq  = (const float*)d_in[2];  const float* bq  = (const float*)d_in[3];
  const float* Wk  = (const float*)d_in[4];  const float* bk  = (const float*)d_in[5];
  const float* Wv  = (const float*)d_in[6];  const float* bv  = (const float*)d_in[7];
  const float* Wg1 = (const float*)d_in[8];  const float* bg1 = (const float*)d_in[9];
  const float* Wg2 = (const float*)d_in[10]; const float* bg2 = (const float*)d_in[11];
  const float* Wo  = (const float*)d_in[12]; const float* bo  = (const float*)d_in[13];

  float* out    = (float*)d_out;
  float* entOut = out + (size_t)BATCH * NSEQ * DMODEL;

  uint8_t* ws = (uint8_t*)d_ws;
  size_t off = 0;
  auto alloc = [&](size_t bytes) -> void* {
    void* p = ws + off;
    off += (bytes + 255) & ~(size_t)255;
    return p;
  };
  const size_t XE = (size_t)MROWS * DMODEL;        // 3.1M elems
  const size_t WE = (size_t)DMODEL * DMODEL;       // 1M elems
  __bf16* xb   = (__bf16*)alloc(XE * 2);
  __bf16* WqT  = (__bf16*)alloc(WE * 2);
  __bf16* WkT  = (__bf16*)alloc(WE * 2);
  __bf16* WvT  = (__bf16*)alloc(WE * 2);
  __bf16* Wg1T = (__bf16*)alloc(WE * 2);
  __bf16* WoT  = (__bf16*)alloc(WE * 2);
  __bf16* Qb   = (__bf16*)alloc(XE * 2);
  __bf16* Kb   = (__bf16*)alloc(XE * 2);
  __bf16* Vt   = (__bf16*)alloc(XE * 2);
  __bf16* Oc   = (__bf16*)alloc(XE * 2);
  float*  gacc = (float*)alloc(MROWS * 4);
  float*  gate = (float*)alloc(MROWS * 4);

  // 1) convert inputs / transpose weights to bf16
  cvt_f32_to_bf16<<<dim3((unsigned)(XE / 256)), 256, 0, stream>>>(x, xb, XE);
  const unsigned tg = (unsigned)(WE / 256);
  transpose_w_bf16<<<tg, 256, 0, stream>>>(Wq,  WqT);
  transpose_w_bf16<<<tg, 256, 0, stream>>>(Wk,  WkT);
  transpose_w_bf16<<<tg, 256, 0, stream>>>(Wv,  WvT);
  transpose_w_bf16<<<tg, 256, 0, stream>>>(Wg1, Wg1T);
  transpose_w_bf16<<<tg, 256, 0, stream>>>(Wo,  WoT);
  zero_f32<<<MROWS / 256, 256, 0, stream>>>(gacc, MROWS);

  // 2) projections (WMMA GEMMs): block tile 64(M) x 128(N)
  dim3 ggrid(DMODEL / 128, MROWS / 64);
  gemm_bf16<0><<<ggrid, 256, 0, stream>>>(xb, WqT,  bq,  (void*)Qb, nullptr, nullptr, DMODEL);
  gemm_bf16<0><<<ggrid, 256, 0, stream>>>(xb, WkT,  bk,  (void*)Kb, nullptr, nullptr, DMODEL);
  gemm_bf16<1><<<ggrid, 256, 0, stream>>>(xb, WvT,  bv,  (void*)Vt, nullptr, nullptr, DMODEL);
  gemm_bf16<2><<<ggrid, 256, 0, stream>>>(xb, Wg1T, bg1, nullptr,   Wg2,     gacc,    DMODEL);
  gate_finalize<<<MROWS / 256, 256, 0, stream>>>(gacc, bg2, gate, MROWS);

  // 3) causal attention + entropy + gate scaling
  dim3 agrid(NSEQ / 16 / 4, BATCH * HEADS);
  attn_kernel<<<agrid, 128, 0, stream>>>(Qb, Kb, Vt, gate, Oc, entOut);

  // 4) output projection -> d_out (fp32)
  gemm_bf16<3><<<ggrid, 256, 0, stream>>>(Oc, WoT, bo, (void*)out, nullptr, nullptr, DMODEL);
}
